// MultiHeadSelfAttention_69664369541658
// MI455X (gfx1250) — matrix operand order repaired and checked by reading, not compiled, not run on hardware
//
#include <hip/hip_runtime.h>

typedef __attribute__((ext_vector_type(16))) _Float16     v16h;
typedef __attribute__((ext_vector_type(8)))  float        v8f;
typedef __attribute__((ext_vector_type(4)))  unsigned int v4u;
typedef __attribute__((ext_vector_type(8)))  int          v8i;
typedef __attribute__((ext_vector_type(4)))  int          v4i;

#define D_MODEL 768
#define N_HEADS 12
#define D_K     64
#define SEQ     4096
#define BATCH   2
#define M_TOTAL (BATCH * SEQ)   // 8192
#define N_QKV   (3 * D_MODEL)   // 2304
#define LDT     40              // padded LDS stride (halves)

static __device__ __forceinline__ v8f wmma_f16(v16h a, v16h b, v8f c) {
    // D = A(16x32 f16) * B(32x16 f16) + C(16x16 f32)
    return __builtin_amdgcn_wmma_f32_16x16x32_f16(
        /*neg_a=*/false, a, /*neg_b=*/false, b,
        /*c_mod=*/(short)0, c, /*reuse_a=*/false, /*reuse_b=*/false);
}

// A-fragment loader (ISA 7.12.2, 16-bit A 16x32):
// lane L: row = L%16 ; elements i<8 -> K = 8*(L/16)+i ; i>=8 -> K = 16+8*(L/16)+(i-8)
static __device__ __forceinline__ v16h load_a_frag(const _Float16* __restrict__ base, int stride) {
    const int lane = threadIdx.x & 31;
    const _Float16* p = base + (lane & 15) * stride + ((lane >> 4) << 3);
    v16h f;
#pragma unroll
    for (int i = 0; i < 8; ++i) f[i] = p[i];
#pragma unroll
    for (int i = 0; i < 8; ++i) f[i + 8] = p[16 + i];
    return f;
}

// B-fragment loader (dense analogue of sparse-B layout, 32x16):
// lane L: col = L%16 ; elements i<8 -> K = 8*(L/16)+i ; i>=8 -> K = 16+8*(L/16)+(i-8). Source is [col][k] row-major (B^T).
static __device__ __forceinline__ v16h load_b_frag(const _Float16* __restrict__ base, int stride) {
    const int lane = threadIdx.x & 31;
    const _Float16* p = base + (lane & 15) * stride + ((lane >> 4) << 3);
    v16h f;
#pragma unroll
    for (int i = 0; i < 8; ++i) f[i] = p[i];
#pragma unroll
    for (int i = 0; i < 8; ++i) f[i + 8] = p[16 + i];
    return f;
}

// ---------------------------------------------------------------------------
// Tensor Data Mover: async 2D-tile load Global -> LDS (ISA ch.8 D# layout).
// 2-byte elements; groups 2/3 zero (2D tensor). Issued by one wave; TENSORcnt.
// ---------------------------------------------------------------------------
static __device__ __forceinline__ void tdm_load_tile_2d(
    unsigned lds_addr, const void* gptr,
    unsigned tensor_d0, unsigned tensor_d1,
    unsigned tile_d0, unsigned tile_d1,
    unsigned long long d0_stride)
{
    unsigned long long ga = (unsigned long long)(uintptr_t)gptr;
    v4u g0;
    g0[0] = 1u;                                    // count=1, user D#, no gather
    g0[1] = lds_addr;                              // bits [63:32]  lds_addr
    g0[2] = (unsigned)(ga & 0xFFFFFFFFu);          // bits [95:64]  global_addr lo
    g0[3] = (unsigned)((ga >> 32) & 0x01FFFFFFu)   // bits [120:96] global_addr hi
          | (2u << 30);                            // bits [127:126] type=2 (image)
    v8i g1;
    g1[0] = (int)(1u << 16);                       // wg_mask=0, data_size=1 (2B)
    g1[1] = (int)((tensor_d0 & 0xFFFFu) << 16);    // barrier_addr=0 | tdim0[15:0]
    g1[2] = (int)(((tensor_d0 >> 16) & 0xFFFFu) | ((tensor_d1 & 0xFFFFu) << 16));
    g1[3] = (int)(((tensor_d1 >> 16) & 0xFFFFu) | ((tile_d0 & 0xFFFFu) << 16));
    g1[4] = (int)(tile_d1 & 0xFFFFu);              // tile_dim1 | tile_dim2=0
    g1[5] = (int)(d0_stride & 0xFFFFFFFFull);      // dim0_stride[31:0]
    g1[6] = (int)((d0_stride >> 32) & 0xFFFFull);  // dim0_stride[47:32] | d1s[15:0]=0
    g1[7] = 0;                                     // dim1_stride[47:16]=0
    v4i z4 = {0, 0, 0, 0};
#if defined(__clang_major__) && (__clang_major__ >= 23)
    v8i z8 = {0, 0, 0, 0, 0, 0, 0, 0};
    __builtin_amdgcn_tensor_load_to_lds(g0, g1, z4, z4, z8, 0);
#else
    __builtin_amdgcn_tensor_load_to_lds(g0, g1, z4, z4, 0);
#endif
}

// ---------------------------------------------------------------------------
// Kernel 1: QKV projection.  C[8192,2304] = X[8192,768] @ Wqkv + b.
// Block = 8 waves, tile 128(M) x 64(N), K-step 32.
// Q,K stored [B,H,S,64] f16 ; V stored transposed [B,H,64,S] f16.
// ---------------------------------------------------------------------------
__global__ __launch_bounds__(256) void qkv_proj_kernel(
    const float* __restrict__ x, const float* __restrict__ w,
    const float* __restrict__ bias,
    _Float16* __restrict__ qh, _Float16* __restrict__ kh, _Float16* __restrict__ vh)
{
    __shared__ _Float16 sA[128 * LDT];
    __shared__ _Float16 sB[64 * LDT];   // W tile transposed: [n][k]
    const int tid  = threadIdx.x;
    const int wave = tid >> 5;
    const int lane = tid & 31;
    const int m0 = blockIdx.y * 128;
    const int n0 = blockIdx.x * 64;

    v8f acc[4];
#pragma unroll
    for (int j = 0; j < 4; ++j)
#pragma unroll
        for (int i = 0; i < 8; ++i) acc[j][i] = 0.0f;

    float bcol[4];
#pragma unroll
    for (int j = 0; j < 4; ++j) bcol[j] = bias[n0 + j * 16 + (lane & 15)];

    for (int kk = 0; kk < D_MODEL; kk += 32) {
        for (int i = tid; i < 128 * 32; i += 256) {
            int r = i >> 5, c = i & 31;
            sA[r * LDT + c] = (_Float16)x[(size_t)(m0 + r) * D_MODEL + kk + c];
        }
        for (int i = tid; i < 32 * 64; i += 256) {
            int r = i >> 6, c = i & 63;           // r = k, c = n
            sB[c * LDT + r] = (_Float16)w[(size_t)(kk + r) * N_QKV + n0 + c];
        }
        __syncthreads();
        v16h a = load_a_frag(sA + wave * 16 * LDT, LDT);
#pragma unroll
        for (int j = 0; j < 4; ++j) {
            v16h b = load_b_frag(sB + j * 16 * LDT, LDT);
            acc[j] = wmma_f16(a, b, acc[j]);
        }
        __syncthreads();
    }

    const int lh = lane >> 4;                 // lane half
    const int nl = lane & 15;
    const int sel = n0 / D_MODEL;             // uniform per block (768 % 64 == 0)
    const int e0  = n0 - sel * D_MODEL;       // 64-aligned -> head uniform per block
    const int h   = e0 >> 6;
#pragma unroll
    for (int j = 0; j < 4; ++j) {
#pragma unroll
        for (int v = 0; v < 8; ++v) {
            int m  = m0 + wave * 16 + v + 8 * lh;   // C/D layout: VGPR v, lane half -> M
            int bi = m >> 12;                        // / 4096
            int s  = m & 4095;
            int d  = j * 16 + nl;
            _Float16 hv = (_Float16)(acc[j][v] + bcol[j]);
            size_t head = (size_t)(bi * N_HEADS + h);
            if (sel == 0)      qh[(head * SEQ + s) * D_K + d] = hv;
            else if (sel == 1) kh[(head * SEQ + s) * D_K + d] = hv;
            else               vh[(head * D_K + d) * SEQ + s] = hv;  // V^T
        }
    }
}

// ---------------------------------------------------------------------------
// Kernel 2: flash attention.  One wave = 16 query rows; key tiles of 32.
// K/V tiles are DMA'd to LDS by the Tensor Data Mover (double-buffered);
// wave 0 issues descriptors and waits on TENSORcnt, block barrier publishes.
// ---------------------------------------------------------------------------
__global__ __launch_bounds__(256) void flash_attn_kernel(
    const _Float16* __restrict__ qh, const _Float16* __restrict__ kh,
    const _Float16* __restrict__ vh, _Float16* __restrict__ ctxh)
{
    __shared__ _Float16 sK[2][32 * 64];      // K tile  [t][d], 4 KB each
    __shared__ _Float16 sV[2][64 * 32];      // V^T tile [d][t], 4 KB each
    __shared__ _Float16 pbuf[8 * 16 * 32];   // per-wave 16x32 P tile (1 KB each)
    const int tid  = threadIdx.x;
    const int wave = tid >> 5;
    const int lane = tid & 31;
    const int lh   = lane >> 4;
    const int nl   = lane & 15;
    const int bh = blockIdx.y;               // b*H + h
    const int bi = bh / N_HEADS;
    const int h  = bh - bi * N_HEADS;
    const int qrow0 = blockIdx.x * 128 + wave * 16;

    const _Float16* qbase = qh + ((size_t)bh * SEQ + qrow0) * D_K;
    const _Float16* kbase = kh + (size_t)bh * SEQ * D_K;
    const _Float16* vbase = vh + (size_t)bh * D_K * SEQ;
    _Float16* pb = pbuf + wave * (16 * 32);

    v16h q0 = load_a_frag(qbase, D_K);        // d = 0..31
    v16h q1 = load_a_frag(qbase + 32, D_K);   // d = 32..63

    v8f acc[4];
    float mrow[8], lrow[8];
#pragma unroll
    for (int v = 0; v < 8; ++v) {
        mrow[v] = -1.0e30f;
        lrow[v] = 0.0f;
#pragma unroll
        for (int j = 0; j < 4; ++j) acc[j][v] = 0.0f;
    }
    const float scale = 0.125f;               // 1/sqrt(64)

    // Prologue: TDM-load tile 0 into buffer 0.
    if (wave == 0) {
        tdm_load_tile_2d((unsigned)(uintptr_t)&sK[0][0], kbase,
                         D_K, SEQ, D_K, 32, D_K);
        tdm_load_tile_2d((unsigned)(uintptr_t)&sV[0][0], vbase,
                         SEQ, D_K, 32, D_K, SEQ);
    }

    for (int t0 = 0; t0 < SEQ; t0 += 32) {
        const int buf = (t0 >> 5) & 1;
        if (wave == 0) {
            if (t0 + 32 < SEQ) {   // prefetch next tile, then wait for current
                tdm_load_tile_2d((unsigned)(uintptr_t)&sK[buf ^ 1][0],
                                 kbase + (size_t)(t0 + 32) * D_K,
                                 D_K, SEQ, D_K, 32, D_K);
                tdm_load_tile_2d((unsigned)(uintptr_t)&sV[buf ^ 1][0],
                                 vbase + (t0 + 32),
                                 SEQ, D_K, 32, D_K, SEQ);
                __builtin_amdgcn_s_wait_tensorcnt(2);
            } else {
                __builtin_amdgcn_s_wait_tensorcnt(0);
            }
        }
        __syncthreads();                       // publish LDS tile to all waves
        const _Float16* sKb = &sK[buf][0];
        const _Float16* sVb = &sV[buf][0];

        v8f c0, c1;
#pragma unroll
        for (int v = 0; v < 8; ++v) { c0[v] = 0.0f; c1[v] = 0.0f; }
        {   // n-tile t0..t0+15
            v16h k0 = load_b_frag(sKb, D_K);
            v16h k1 = load_b_frag(sKb + 32, D_K);
            c0 = wmma_f16(q0, k0, c0);
            c0 = wmma_f16(q1, k1, c0);
        }
        {   // n-tile t0+16..t0+31
            v16h k0 = load_b_frag(sKb + 16 * D_K, D_K);
            v16h k1 = load_b_frag(sKb + 16 * D_K + 32, D_K);
            c1 = wmma_f16(q0, k0, c1);
            c1 = wmma_f16(q1, k1, c1);
        }
        // Online softmax in fragment layout: lane = N, VGPR v + lane-half = M.
#pragma unroll
        for (int v = 0; v < 8; ++v) {
            float s0 = c0[v] * scale;
            float s1 = c1[v] * scale;
            float mx = fmaxf(s0, s1);
            mx = fmaxf(mx, __shfl_xor(mx, 1));
            mx = fmaxf(mx, __shfl_xor(mx, 2));
            mx = fmaxf(mx, __shfl_xor(mx, 4));
            mx = fmaxf(mx, __shfl_xor(mx, 8));   // row max within 16-lane half
            float mn   = fmaxf(mrow[v], mx);
            float corr = __expf(mrow[v] - mn);
            mrow[v] = mn;
            float p0 = __expf(s0 - mn);
            float p1 = __expf(s1 - mn);
            float rs = p0 + p1;
            rs += __shfl_xor(rs, 1);
            rs += __shfl_xor(rs, 2);
            rs += __shfl_xor(rs, 4);
            rs += __shfl_xor(rs, 8);
            lrow[v] = lrow[v] * corr + rs;
#pragma unroll
            for (int j = 0; j < 4; ++j) acc[j][v] *= corr;
            int m = v + 8 * lh;
            pb[m * 32 + nl]      = (_Float16)p0;   // D-layout -> row-major P in LDS
            pb[m * 32 + 16 + nl] = (_Float16)p1;
        }
        __asm__ volatile("" ::: "memory");
        __builtin_amdgcn_wave_barrier();           // wave-private LDS transpose
        v16h pa = load_a_frag(pb, 32);             // P as A-fragment (16x32)
#pragma unroll
        for (int j = 0; j < 4; ++j) {
            v16h vf = load_b_frag(sVb + (j * 16) * 32, 32);
            acc[j] = wmma_f16(pa, vf, acc[j]);
        }
        __asm__ volatile("" ::: "memory");
        __builtin_amdgcn_wave_barrier();
        __syncthreads();                       // tile consumed; safe to refill
    }

#pragma unroll
    for (int j = 0; j < 4; ++j) {
#pragma unroll
        for (int v = 0; v < 8; ++v) {
            int s = qrow0 + v + 8 * lh;
            ctxh[((size_t)bi * SEQ + s) * D_MODEL + h * D_K + j * 16 + nl] =
                (_Float16)(acc[j][v] / lrow[v]);
        }
    }
}

// ---------------------------------------------------------------------------
// Kernel 3: output projection.  out[8192,768] = ctx_f16 @ Wout + b (f32 out).
// ---------------------------------------------------------------------------
__global__ __launch_bounds__(256) void out_proj_kernel(
    const _Float16* __restrict__ ctxh, const float* __restrict__ w,
    const float* __restrict__ bias, float* __restrict__ out)
{
    __shared__ _Float16 sA[128 * LDT];
    __shared__ _Float16 sB[64 * LDT];
    const int tid  = threadIdx.x;
    const int wave = tid >> 5;
    const int lane = tid & 31;
    const int m0 = blockIdx.y * 128;
    const int n0 = blockIdx.x * 64;

    v8f acc[4];
#pragma unroll
    for (int j = 0; j < 4; ++j)
#pragma unroll
        for (int i = 0; i < 8; ++i) acc[j][i] = 0.0f;

    float bcol[4];
#pragma unroll
    for (int j = 0; j < 4; ++j) bcol[j] = bias[n0 + j * 16 + (lane & 15)];

    for (int kk = 0; kk < D_MODEL; kk += 32) {
        for (int i = tid; i < 128 * 32; i += 256) {
            int r = i >> 5, c = i & 31;
            sA[r * LDT + c] = ctxh[(size_t)(m0 + r) * D_MODEL + kk + c];
        }
        for (int i = tid; i < 32 * 64; i += 256) {
            int r = i >> 6, c = i & 63;
            sB[c * LDT + r] = (_Float16)w[(size_t)(kk + r) * D_MODEL + n0 + c];
        }
        __syncthreads();
        v16h a = load_a_frag(sA + wave * 16 * LDT, LDT);
#pragma unroll
        for (int j = 0; j < 4; ++j) {
            v16h b = load_b_frag(sB + j * 16 * LDT, LDT);
            acc[j] = wmma_f16(a, b, acc[j]);
        }
        __syncthreads();
    }

    const int lh = lane >> 4;
    const int nl = lane & 15;
#pragma unroll
    for (int j = 0; j < 4; ++j) {
#pragma unroll
        for (int v = 0; v < 8; ++v) {
            int m = m0 + wave * 16 + v + 8 * lh;
            out[(size_t)m * D_MODEL + n0 + j * 16 + nl] = acc[j][v] + bcol[j];
        }
    }
}

extern "C" void kernel_launch(void* const* d_in, const int* in_sizes, int n_in,
                              void* d_out, int out_size, void* d_ws, size_t ws_size,
                              hipStream_t stream) {
    (void)in_sizes; (void)n_in; (void)out_size; (void)ws_size;
    const float* x     = (const float*)d_in[0];
    const float* w_qkv = (const float*)d_in[1];
    const float* b_qkv = (const float*)d_in[2];
    const float* w_out = (const float*)d_in[3];
    const float* b_out = (const float*)d_in[4];
    float* out = (float*)d_out;

    // Workspace layout (f16): Q,K [B,H,S,64] ; V^T [B,H,64,S] ; ctx [B,S,768]
    const size_t QKV_ELEMS = (size_t)BATCH * N_HEADS * SEQ * D_K;  // 6,291,456
    _Float16* qh   = (_Float16*)d_ws;
    _Float16* kh   = qh + QKV_ELEMS;
    _Float16* vh   = kh + QKV_ELEMS;
    _Float16* ctxh = vh + QKV_ELEMS;   // total ~50.3 MB

    dim3 blk(256);
    dim3 g1(N_QKV / 64, M_TOTAL / 128);       // 36 x 64
    qkv_proj_kernel<<<g1, blk, 0, stream>>>(x, w_qkv, b_qkv, qh, kh, vh);

    dim3 g2(SEQ / 128, BATCH * N_HEADS);      // 32 x 24
    flash_attn_kernel<<<g2, blk, 0, stream>>>(qh, kh, vh, ctxh);

    dim3 g3(D_MODEL / 64, M_TOTAL / 128);     // 12 x 64
    out_proj_kernel<<<g3, blk, 0, stream>>>(ctxh, w_out, b_out, out);
}
